// TransformerImputer_82240033784119
// MI455X (gfx1250) — matrix operand order repaired and checked by reading, not compiled, not run on hardware
//
#include <hip/hip_runtime.h>
#include <hip/hip_bf16.h>
#include <math.h>

typedef __bf16 bf16_t;
typedef __attribute__((ext_vector_type(16))) __bf16 v16bf;
typedef __attribute__((ext_vector_type(8)))  __bf16 v8bf;
typedef __attribute__((ext_vector_type(8)))  float  v8f;

#define GT_NONE 0
#define GT_RELU 1
#define GT_GELU 2
#define GT_SOFTPLUS 3

#define CAT16(a, b) __builtin_shufflevector(a, b, 0,1,2,3,4,5,6,7,8,9,10,11,12,13,14,15)

static __host__ __device__ inline int cdiv_i(int a, int b) { return (a + b - 1) / b; }

// ---------------------------------------------------------------------------
// Fused WMMA GEMM: C[M,N] = act(A[M,K] @ W[K,N] + bias + resid)
// A fp32 row-major (lda=K). Wt = weight TRANSPOSED + padded: [Npad rows = n][K],
// rows n >= N are zero, so B loads need no guards (Npad multiple of 128).
// Optional conv row-shift: if Tp>0, A row r reads r+shiftT iff (r%Tp)+shiftT
// in [0,Tp), else zeros.
// Block: 256 thr = 8 waves (2x4); tile 64(M) x 128(N); K-step 32.
// Each wave owns a 2x2 grid of 16x16 accumulators -> 4 WMMA per K-step
// against 2 A-fragments (LDS) + 2 B-fragments (global, L2-resident).
// ---------------------------------------------------------------------------
__global__ __launch_bounds__(256) void gemm_wmma_bf16(
    const float* __restrict__ A,
    const bf16_t* __restrict__ Wt,
    const float* __restrict__ bias,
    const float* __restrict__ resid,
    float* __restrict__ C,
    int M, int N, int K, int act, int shiftT, int Tp)
{
    __shared__ bf16_t sA[64][40];

    const int tid  = threadIdx.x;
    const int bm   = blockIdx.y, bn = blockIdx.x;
    const int wave = tid >> 5, l = tid & 31;
    const int half = l >> 4, lr = l & 15;
    const int wm   = wave & 1, wn = wave >> 1;   // 2 x 4 wave grid

    const v8f z8 = {0.f,0.f,0.f,0.f,0.f,0.f,0.f,0.f};
    v8f acc00 = z8, acc01 = z8, acc10 = z8, acc11 = z8;

    // A staging map: 64 rows x 32 k, 8 consecutive k per thread (2x float4)
    const int arow = tid >> 2, ak0 = (tid & 3) * 8;
    int gmA = bm * 64 + arow;
    bool rowValid = (gmA < M);
    long srcRow = gmA;
    if (Tp) {
        int t = gmA % Tp;
        int ts = t + shiftT;
        if (ts < 0 || ts >= Tp) rowValid = false;
        srcRow = (long)gmA + shiftT;
    }
    // B: two output-column rows per wave-lane (contiguous in k, zero-padded)
    const bf16_t* wrow0 = Wt + (long)(bn * 128 + wn * 32 + lr) * K;
    const bf16_t* wrow1 = Wt + (long)(bn * 128 + wn * 32 + 16 + lr) * K;

    for (int k0 = 0; k0 < K; k0 += 32) {
        if (rowValid) {
            const float4 v4a = *(const float4*)(A + srcRow * (long)K + k0 + ak0);
            const float4 v4b = *(const float4*)(A + srcRow * (long)K + k0 + ak0 + 4);
            union { bf16_t h[8]; uint4 u; } pk;
            pk.h[0] = (bf16_t)v4a.x; pk.h[1] = (bf16_t)v4a.y;
            pk.h[2] = (bf16_t)v4a.z; pk.h[3] = (bf16_t)v4a.w;
            pk.h[4] = (bf16_t)v4b.x; pk.h[5] = (bf16_t)v4b.y;
            pk.h[6] = (bf16_t)v4b.z; pk.h[7] = (bf16_t)v4b.w;
            *(uint4*)&sA[arow][ak0] = pk.u;
        } else {
            *(uint4*)&sA[arow][ak0] = make_uint4(0u, 0u, 0u, 0u);
        }
        __syncthreads();

        if (rowValid && (k0 + 32 < K))
            __builtin_prefetch(A + srcRow * (long)K + k0 + 32 + ak0, 0, 3);

        // A fragments (two 16-row subtiles): contiguous 8-elem LDS chunks
        const int r0 = wm * 32 + lr;
        v16bf af0 = CAT16(*(const v8bf*)&sA[r0][half * 8],
                          *(const v8bf*)&sA[r0][16 + half * 8]);
        v16bf af1 = CAT16(*(const v8bf*)&sA[r0 + 16][half * 8],
                          *(const v8bf*)&sA[r0 + 16][16 + half * 8]);
        // B fragments (two 16-col subtiles): contiguous global b128 pairs
        const v8bf* wp0 = (const v8bf*)(wrow0 + k0 + half * 8);
        const v8bf* wp1 = (const v8bf*)(wrow1 + k0 + half * 8);
        v16bf bf0 = CAT16(wp0[0], wp0[2]);
        v16bf bf1 = CAT16(wp1[0], wp1[2]);

        acc00 = __builtin_amdgcn_wmma_f32_16x16x32_bf16(false, af0, false, bf0, (short)0, acc00, false, false);
        acc01 = __builtin_amdgcn_wmma_f32_16x16x32_bf16(false, af0, false, bf1, (short)0, acc01, false, false);
        acc10 = __builtin_amdgcn_wmma_f32_16x16x32_bf16(false, af1, false, bf0, (short)0, acc10, false, false);
        acc11 = __builtin_amdgcn_wmma_f32_16x16x32_bf16(false, af1, false, bf1, (short)0, acc11, false, false);
        __syncthreads();
    }

    const v8f* accs[2][2] = { {&acc00, &acc01}, {&acc10, &acc11} };
    #pragma unroll
    for (int i = 0; i < 2; ++i) {
        #pragma unroll
        for (int j = 0; j < 2; ++j) {
            const v8f a = *accs[i][j];
            #pragma unroll
            for (int r = 0; r < 8; ++r) {
                int gm = bm * 64 + wm * 32 + i * 16 + half * 8 + r;
                int gn = bn * 128 + wn * 32 + j * 16 + lr;
                if (gm < M && gn < N) {
                    float v = a[r];
                    if (bias)  v += bias[gn];
                    if (resid) v += resid[(long)gm * N + gn];
                    if (act == GT_RELU)          v = fmaxf(v, 0.f);
                    else if (act == GT_GELU)     v = 0.5f * v * (1.f + erff(v * 0.70710678118f));
                    else if (act == GT_SOFTPLUS) v = (v > 20.f) ? v : log1pf(expf(v));
                    C[(long)gm * N + gn] = v;
                }
            }
        }
    }
}

// ---------------------------------------------------------------------------
// Fused attention. Inputs pre-packed in bf16:
//   qk : (B*T) x 1024 row-major  [q(512) | k(512)]
//   vt : (B,H,64,1024)           V transposed per head, cols s>=T zero-padded
// Per (b,h,16-row t block): S=QK^T (WMMA) + rel bias + mask, softmax in LDS,
// O = P@V (WMMA) with 1/sum epilogue. 128 thr = 4 waves. All fragment loads
// are pairs of contiguous b128; out-of-range rows clamped (scores overwritten).
// ---------------------------------------------------------------------------
#define ATT_TPAD 1024
__global__ __launch_bounds__(128) void attn_wmma(
    const bf16_t* __restrict__ qk,
    const bf16_t* __restrict__ vt,
    const float* __restrict__ rel,      // (2T-1) x H
    const float* __restrict__ mbias,    // H
    const int*   __restrict__ mask_ts,  // B*T
    float* __restrict__ out,            // (B*T) x 512
    int B_, int T_, int H_)
{
    __shared__ bf16_t sS[16][ATT_TPAD];
    __shared__ float  redbuf[16][8];
    __shared__ float  rowmax[16];
    __shared__ float  rowsum[16];

    const int b  = blockIdx.y / H_, h = blockIdx.y % H_;
    const int t0 = blockIdx.x * 16;
    const int tid  = threadIdx.x;
    const int wave = tid >> 5, l = tid & 31;
    const int half = l >> 4, lr = l & 15;
    const long base = (long)b * T_;

    // --- Q fragments (dk=64 -> two K=32 steps); row clamped, no branches
    v16bf qf[2];
    {
        int t = t0 + lr; if (t >= T_) t = T_ - 1;
        const bf16_t* qrow = qk + (base + t) * 1024 + h * 64;
        #pragma unroll
        for (int ks = 0; ks < 2; ++ks) {
            v8bf q0 = *(const v8bf*)(qrow + ks * 32 + half * 8);
            v8bf q1 = *(const v8bf*)(qrow + ks * 32 + 16 + half * 8);
            qf[ks] = CAT16(q0, q1);
        }
    }

    // --- scores: each wave owns s-tiles wave, wave+4, ... (Q reused throughout)
    for (int st = wave; st < ATT_TPAD / 16; st += 4) {
        int s0 = st * 16;
        if (s0 >= T_) {                       // pure padding tile (wave-uniform)
            #pragma unroll
            for (int r = 0; r < 8; ++r)
                sS[half * 8 + r][s0 + lr] = (bf16_t)(-INFINITY);
            continue;
        }
        int s = s0 + lr;
        int sc_ = (s < T_) ? s : (T_ - 1);
        const bf16_t* krow = qk + (base + sc_) * 1024 + 512 + h * 64;
        v8f acc = {0.f,0.f,0.f,0.f,0.f,0.f,0.f,0.f};
        #pragma unroll
        for (int ks = 0; ks < 2; ++ks) {
            const v8bf* kp = (const v8bf*)(krow + ks * 32 + half * 8);
            v16bf kf = CAT16(kp[0], kp[2]);
            acc = __builtin_amdgcn_wmma_f32_16x16x32_bf16(
                false, qf[ks], false, kf, (short)0, acc, false, false);
        }
        #pragma unroll
        for (int r = 0; r < 8; ++r) {
            int m = half * 8 + r;
            int t = t0 + m;
            float v;
            if (s >= T_) {
                v = -INFINITY;
            } else {
                v = acc[r] * 0.125f;                       // 1/sqrt(64)
                int tc = (t < T_) ? t : (T_ - 1);
                int i1 = tc - s + (T_ - 1);
                int i2 = s - tc + (T_ - 1);
                v += 0.5f * (rel[i1 * H_ + h] - rel[i2 * H_ + h]);
                if (!mask_ts[base + s]) v = -1e30f + mbias[h];
            }
            sS[m][s0 + lr] = (bf16_t)v;
        }
    }
    __syncthreads();

    // --- softmax over 1024 cols: 8 threads per row
    {
        int row = tid >> 3, j = tid & 7;
        float mx = -INFINITY;
        for (int c = j; c < ATT_TPAD; c += 8) mx = fmaxf(mx, (float)sS[row][c]);
        redbuf[row][j] = mx;
        __syncthreads();
        if (j == 0) {
            float m2 = -INFINITY;
            for (int q = 0; q < 8; ++q) m2 = fmaxf(m2, redbuf[row][q]);
            rowmax[row] = m2;
        }
        __syncthreads();
        float mrow = rowmax[row];
        float sum = 0.f;
        for (int c = j; c < ATT_TPAD; c += 8) {
            float p = expf((float)sS[row][c] - mrow);
            sS[row][c] = (bf16_t)p;                        // un-normalized prob
            sum += p;
        }
        redbuf[row][j] = sum;
        __syncthreads();
        if (j == 0) {
            float s2 = 0.f;
            for (int q = 0; q < 8; ++q) s2 += redbuf[row][q];
            rowsum[row] = s2;
        }
        __syncthreads();
    }

    // --- O = P @ V ; each wave owns one 16-wide d tile; V padded -> no guards
    {
        const int d0 = wave * 16;
        const bf16_t* vrow = vt + (((long)b * H_ + h) * 64 + d0 + lr) * ATT_TPAD;
        v8f acc = {0.f,0.f,0.f,0.f,0.f,0.f,0.f,0.f};
        for (int ks = 0; ks < ATT_TPAD / 32; ++ks) {
            v8bf p0 = *(const v8bf*)&sS[lr][ks * 32 + half * 8];
            v8bf p1 = *(const v8bf*)&sS[lr][ks * 32 + 16 + half * 8];
            v16bf pf = CAT16(p0, p1);
            const v8bf* vp = (const v8bf*)(vrow + ks * 32 + half * 8);
            v16bf vf = CAT16(vp[0], vp[2]);
            acc = __builtin_amdgcn_wmma_f32_16x16x32_bf16(
                false, pf, false, vf, (short)0, acc, false, false);
        }
        #pragma unroll
        for (int r = 0; r < 8; ++r) {
            int m = half * 8 + r, t = t0 + m;
            if (t < T_)
                out[(base + t) * 512 + h * 64 + d0 + lr] = acc[r] / rowsum[m];
        }
    }
}

// ---------------------------------------------------------------------------
// LayerNorm: one row (D=512) per block of 256 threads.
// ---------------------------------------------------------------------------
__global__ __launch_bounds__(256) void layernorm_k(
    const float* __restrict__ x, const float* __restrict__ g,
    const float* __restrict__ bt, float* __restrict__ y)
{
    __shared__ float red[8], red2[8], stats[2];
    const long row = blockIdx.x;
    const float* xr = x + row * 512;
    const int tid = threadIdx.x;
    float v0 = xr[tid], v1 = xr[tid + 256];
    float s = v0 + v1, ss = v0 * v0 + v1 * v1;
    for (int off = 16; off; off >>= 1) {
        s  += __shfl_down(s, off, 32);
        ss += __shfl_down(ss, off, 32);
    }
    if ((tid & 31) == 0) { red[tid >> 5] = s; red2[tid >> 5] = ss; }
    __syncthreads();
    if (tid == 0) {
        float a = 0.f, b2 = 0.f;
        for (int i = 0; i < 8; ++i) { a += red[i]; b2 += red2[i]; }
        float mu = a / 512.f;
        float var = b2 / 512.f - mu * mu;
        stats[0] = mu;
        stats[1] = rsqrtf(fmaxf(var, 0.f) + 1e-5f);
    }
    __syncthreads();
    float mu = stats[0], rs = stats[1];
    y[row * 512 + tid]       = (v0 - mu) * rs * g[tid] + bt[tid];
    y[row * 512 + tid + 256] = (v1 - mu) * rs * g[tid + 256] + bt[tid + 256];
}

// ---------------------------------------------------------------------------
// Conversion / elementwise / setup kernels
// ---------------------------------------------------------------------------
// Transpose + pad: src is K x Nsrc (row-major); dst rows n in [0,Nrows) at
// dst[(rowOff+n)*K + k]; rows with n >= Nsrc are written zero.
__global__ void cvtT_k(const float* __restrict__ src, bf16_t* __restrict__ dst,
                       int K, int Nsrc, int Nrows, int rowOff) {
    long i = (long)blockIdx.x * 256 + threadIdx.x;
    if (i >= (long)Nrows * K) return;
    int n = (int)(i / K), k = (int)(i % K);
    dst[(long)(rowOff + n) * K + k] =
        (n < Nsrc) ? (bf16_t)src[(long)k * Nsrc + n] : (bf16_t)0.f;
}
// Conv weights (O,I,3) -> per-tap transposed [tap][o][i] (rows=o=n, cols=i=k)
__global__ void cvt_convT_k(const float* __restrict__ w, bf16_t* __restrict__ d, int Dc) {
    long i = (long)blockIdx.x * 256 + threadIdx.x;
    if (i >= (long)Dc * Dc * 3) return;
    int o = (int)(i / (Dc * 3)), ii = (int)((i / 3) % Dc), tap = (int)(i % 3);
    d[(long)tap * Dc * Dc + (long)o * Dc + ii] = (bf16_t)w[i];
}
// fp32 qkv rows (1536) -> bf16 qk rows (first 1024 cols: q|k)
__global__ void cvtqk_k(const float* __restrict__ qkv, bf16_t* __restrict__ qk, long rows) {
    long i = (long)blockIdx.x * 256 + threadIdx.x;
    if (i >= rows * 1024) return;
    long r = i >> 10; int c = (int)(i & 1023);
    qk[i] = (bf16_t)qkv[r * 1536 + c];
}
// fp32 qkv v-part -> bf16 vt[b][h][d=64][s=1024], zero-padded past T
__global__ void vtrans_k(const float* __restrict__ qkv, bf16_t* __restrict__ vt,
                         int B_, int T_, int H_) {
    long i = (long)blockIdx.x * 256 + threadIdx.x;
    long n = (long)B_ * H_ * 64 * ATT_TPAD;
    if (i >= n) return;
    int s = (int)(i & (ATT_TPAD - 1));
    int d = (int)((i >> 10) & 63);
    int h = (int)((i >> 16) & 7);
    int b = (int)(i >> 19);
    vt[i] = (s < T_) ? (bf16_t)qkv[((long)b * T_ + s) * 1536 + 1024 + h * 64 + d]
                     : (bf16_t)0.f;
}
__global__ void copyf_k(const float* __restrict__ s, float* __restrict__ d, long n) {
    long i = (long)blockIdx.x * 256 + threadIdx.x;
    if (i < n) d[i] = s[i];
}
__global__ void zero_k(float* p, int n) {
    int i = blockIdx.x * 256 + threadIdx.x;
    if (i < n) p[i] = 0.f;
}
__global__ void buildz_k(const float* __restrict__ X, const int* __restrict__ M,
                         float* __restrict__ z, int rows, int di) {
    long i = (long)blockIdx.x * 256 + threadIdx.x;
    if (i >= (long)rows * di) return;
    int r = (int)(i / di), j = (int)(i % di);
    z[(long)r * 2 * di + j]      = X[i];
    z[(long)r * 2 * di + di + j] = M[i] ? 0.f : 1.f;
}
__global__ void maskts_k(const int* __restrict__ M, int* __restrict__ mts, int rows, int di) {
    int r = blockIdx.x * 256 + threadIdx.x;
    if (r >= rows) return;
    int any = 0;
    for (int j = 0; j < di; ++j) any |= M[(long)r * di + j];
    mts[r] = any ? 1 : 0;
}
__global__ void maskembed_k(float* __restrict__ h, const int* __restrict__ mts,
                            const float* __restrict__ me, int rows) {
    long i = (long)blockIdx.x * 256 + threadIdx.x;
    if (i >= (long)rows * 512) return;
    int r = (int)(i >> 9);
    if (!mts[r]) h[i] = me[i & 511];
}
__global__ void pe_k(float* __restrict__ pe, int T_, int D_) {
    long i = (long)blockIdx.x * 256 + threadIdx.x;
    long np = (long)T_ * (D_ / 2);
    if (i >= np) return;
    int t = (int)(i / (D_ / 2)), p = (int)(i % (D_ / 2));
    float div = expf((float)(2 * p) * (-logf(10000.f) / (float)D_));
    float ang = (float)t * div;
    pe[(long)t * D_ + 2 * p]     = sinf(ang);
    pe[(long)t * D_ + 2 * p + 1] = cosf(ang);
}
__global__ void addpe_k(const float* __restrict__ h, const float* __restrict__ pe,
                        float* __restrict__ x, int T_) {
    long i = (long)blockIdx.x * 256 + threadIdx.x;
    if (i >= 8000L * 512) return;
    int r = (int)(i >> 9), c = (int)(i & 511);
    x[i] = h[i] + pe[(long)(r % T_) * 512 + c];
}
__global__ void add_k(const float* __restrict__ a, const float* __restrict__ b,
                      float* __restrict__ c, long n) {
    long i = (long)blockIdx.x * 256 + threadIdx.x;
    if (i < n) c[i] = a[i] + b[i];
}

// ---------------------------------------------------------------------------
// Global statistics for sigma rescaling
// sc: [0]=n_miss [1]=sum err*miss [2]=sum sqrt(pi/2)|mean|*miss [3]=sum miss*(err-mu)^2
// ---------------------------------------------------------------------------
__global__ __launch_bounds__(256) void red1_k(
    const float* __restrict__ mean, const float* __restrict__ X,
    const int* __restrict__ M, float* __restrict__ sc, int n)
{
    __shared__ float sh[3][8];
    int i = blockIdx.x * 256 + threadIdx.x;
    float c = 0.f, e = 0.f, p = 0.f;
    if (i < n && M[i] == 0) {
        c = 1.f;
        float err = fabsf(mean[i] - X[i]);
        e = err;
        p = 1.2533141373155003f * fabsf(mean[i]);
    }
    for (int off = 16; off; off >>= 1) {
        c += __shfl_down(c, off, 32);
        e += __shfl_down(e, off, 32);
        p += __shfl_down(p, off, 32);
    }
    int w = threadIdx.x >> 5;
    if ((threadIdx.x & 31) == 0) { sh[0][w] = c; sh[1][w] = e; sh[2][w] = p; }
    __syncthreads();
    if (threadIdx.x == 0) {
        float C = 0.f, E = 0.f, P = 0.f;
        for (int q = 0; q < 8; ++q) { C += sh[0][q]; E += sh[1][q]; P += sh[2][q]; }
        atomicAdd(&sc[0], C); atomicAdd(&sc[1], E); atomicAdd(&sc[2], P);
    }
}
__global__ __launch_bounds__(256) void red2_k(
    const float* __restrict__ mean, const float* __restrict__ X,
    const int* __restrict__ M, float* __restrict__ sc, int n)
{
    __shared__ float sh[8];
    float mu_e = sc[1] / sc[0];
    int i = blockIdx.x * 256 + threadIdx.x;
    float v = 0.f;
    if (i < n && M[i] == 0) {
        float err = fabsf(mean[i] - X[i]) - mu_e;
        v = err * err;
    }
    for (int off = 16; off; off >>= 1) v += __shfl_down(v, off, 32);
    if ((threadIdx.x & 31) == 0) sh[threadIdx.x >> 5] = v;
    __syncthreads();
    if (threadIdx.x == 0) {
        float V = 0.f;
        for (int q = 0; q < 8; ++q) V += sh[q];
        atomicAdd(&sc[3], V);
    }
}
__global__ void final_k(const float* __restrict__ mean, const float* __restrict__ alea,
                        const float* __restrict__ X, const int* __restrict__ M,
                        const float* __restrict__ sc, float* __restrict__ out, int n)
{
    int i = blockIdx.x * 256 + threadIdx.x;
    if (i >= n) return;
    float nmiss      = sc[0];
    float target_std = sqrtf(sc[3] / (nmiss - 1.f));
    float pred_mean  = sc[2] / nmiss;
    bool  m  = (M[i] != 0);
    float sg = sqrtf(fmaxf(alea[i], 1e-6f)) * (m ? 1.f : 1.5f) * 1.2f;
    float sig_obs  = sg * 0.4f;
    float sig_miss = 1.2533141373155003f * fabsf(mean[i]) * (target_std / (pred_mean + 1e-6f));
    sig_miss = fmaxf(sig_miss, sg * 0.25f);
    out[i]     = m ? X[i] : mean[i];
    out[n + i] = m ? sig_obs : sig_miss;
}

// ---------------------------------------------------------------------------
// Host orchestration
// ---------------------------------------------------------------------------
static void launch_gemm(const float* A, const bf16_t* Wt, const float* bias,
                        const float* resid, float* C, int M, int N, int K,
                        int act, int shiftT, int Tp, hipStream_t st)
{
    dim3 g(cdiv_i(N, 128), cdiv_i(M, 64));
    gemm_wmma_bf16<<<g, 256, 0, st>>>(A, Wt, bias, resid, C, M, N, K, act, shiftT, Tp);
}

extern "C" void kernel_launch(void* const* d_in, const int* in_sizes, int n_in,
                              void* d_out, int out_size, void* d_ws, size_t ws_size,
                              hipStream_t stream)
{
    (void)in_sizes; (void)n_in; (void)ws_size; (void)out_size;
    const int B_ = 8, T_ = 1000, DI = 32, D = 512, H = 8, L = 3;
    const int R = B_ * T_;              // 8000 rows
    const long RD = (long)R * D;

    // ---- inputs (dict order: X, M, params-leaves) ----
    const float* X = (const float*)d_in[0];
    const int*   M = (const int*)d_in[1];
    const float* embed_w1 = (const float*)d_in[2];
    const float* embed_b1 = (const float*)d_in[3];
    const float* embed_w2 = (const float*)d_in[4];
    const float* embed_b2 = (const float*)d_in[5];
    const float* missing_embed = (const float*)d_in[6];
    const float* lw[L][18];
    for (int li = 0; li < L; ++li)
        for (int j = 0; j < 18; ++j)
            lw[li][j] = (const float*)d_in[7 + li * 18 + j];
    // layer leaves: wq bq wk bk wv bv wo bo missing_bias rel ln1_s ln1_b
    //               ff_w1 ff_b1 ff_w2 ff_b2 ln2_s ln2_b
    const float* conv_w[3]; const float* conv_b[3];
    for (int ci = 0; ci < 3; ++ci) {
        conv_w[ci] = (const float*)d_in[61 + 2 * ci];
        conv_b[ci] = (const float*)d_in[62 + 2 * ci];
    }
    const float* mean_w1 = (const float*)d_in[67];
    const float* mean_b1 = (const float*)d_in[68];
    const float* mean_w2 = (const float*)d_in[69];
    const float* mean_b2 = (const float*)d_in[70];
    const float* alea_w1 = (const float*)d_in[71];
    const float* alea_b1 = (const float*)d_in[72];
    const float* alea_w2 = (const float*)d_in[73];
    const float* alea_b2 = (const float*)d_in[74];
    float* out = (float*)d_out;

    // ---- workspace carve ----
    char* wp = (char*)d_ws;
    auto carve = [&](size_t bytes) -> void* {
        void* p = (void*)wp;
        wp += (bytes + 255) & ~(size_t)255;
        return p;
    };
    // transposed+padded bf16 weights: [Npad rows][K]
    bf16_t* bw_e1   = (bf16_t*)carve((size_t)512 * 64 * 2);    // N=512, K=64
    bf16_t* bw_e2   = (bf16_t*)carve((size_t)512 * 512 * 2);
    bf16_t* bw_qkv[L]; bf16_t* bw_o[L]; bf16_t* bw_f1[L]; bf16_t* bw_f2[L];
    float*  qb[L];
    for (int li = 0; li < L; ++li) {
        bw_qkv[li] = (bf16_t*)carve((size_t)1536 * 512 * 2);
        bw_o[li]   = (bf16_t*)carve((size_t)512 * 512 * 2);
        bw_f1[li]  = (bf16_t*)carve((size_t)2048 * 512 * 2);
        bw_f2[li]  = (bf16_t*)carve((size_t)512 * 2048 * 2);
        qb[li]     = (float*)carve((size_t)1536 * 4);
    }
    bf16_t* bw_conv[3];
    for (int ci = 0; ci < 3; ++ci)
        bw_conv[ci] = (bf16_t*)carve((size_t)3 * 512 * 512 * 2);
    bf16_t* bw_m1 = (bf16_t*)carve((size_t)512 * 512 * 2);
    bf16_t* bw_m2 = (bf16_t*)carve((size_t)128 * 512 * 2);    // N=32 padded to 128
    bf16_t* bw_a1 = (bf16_t*)carve((size_t)512 * 512 * 2);
    bf16_t* bw_a2 = (bf16_t*)carve((size_t)128 * 512 * 2);

    float*  pe    = (float*)carve((size_t)T_ * D * 4);
    float*  z     = (float*)carve((size_t)R * 2 * DI * 4);
    float*  h     = (float*)carve((size_t)RD * 4);
    float*  xb    = (float*)carve((size_t)RD * 4);
    float*  qkv   = (float*)carve((size_t)R * 1536 * 4);
    bf16_t* qkbf  = (bf16_t*)carve((size_t)R * 1024 * 2);
    bf16_t* vtbf  = (bf16_t*)carve((size_t)B_ * H * 64 * ATT_TPAD * 2);
    float*  attn  = (float*)carve((size_t)RD * 4);
    float*  t1    = (float*)carve((size_t)RD * 4);
    float*  ff1   = (float*)carve((size_t)R * 2048 * 4);
    float*  hl    = (float*)carve((size_t)RD * 4);
    float*  hl2   = (float*)carve((size_t)RD * 4);
    float*  meanb = (float*)carve((size_t)R * DI * 4);
    float*  aleab = (float*)carve((size_t)R * DI * 4);
    int*    mts   = (int*)carve((size_t)R * 4);
    float*  sc    = (float*)carve((size_t)8 * 4);

    auto g1 = [](long n) { return dim3((unsigned)((n + 255) / 256)); };

    // ---- setup: scalars, weight transposition, pe, masks, z ----
    zero_k<<<1, 256, 0, stream>>>(sc, 8);
    cvtT_k<<<g1(512L * 64), 256, 0, stream>>>(embed_w1, bw_e1, 64, 512, 512, 0);
    cvtT_k<<<g1(512L * 512), 256, 0, stream>>>(embed_w2, bw_e2, 512, 512, 512, 0);
    for (int li = 0; li < L; ++li) {
        cvtT_k<<<g1(512L * 512), 256, 0, stream>>>(lw[li][0], bw_qkv[li], 512, 512, 512, 0);
        cvtT_k<<<g1(512L * 512), 256, 0, stream>>>(lw[li][2], bw_qkv[li], 512, 512, 512, 512);
        cvtT_k<<<g1(512L * 512), 256, 0, stream>>>(lw[li][4], bw_qkv[li], 512, 512, 512, 1024);
        copyf_k<<<g1(512), 256, 0, stream>>>(lw[li][1], qb[li] + 0, 512);
        copyf_k<<<g1(512), 256, 0, stream>>>(lw[li][3], qb[li] + 512, 512);
        copyf_k<<<g1(512), 256, 0, stream>>>(lw[li][5], qb[li] + 1024, 512);
        cvtT_k<<<g1(512L * 512), 256, 0, stream>>>(lw[li][6], bw_o[li], 512, 512, 512, 0);
        cvtT_k<<<g1(2048L * 512), 256, 0, stream>>>(lw[li][12], bw_f1[li], 512, 2048, 2048, 0);
        cvtT_k<<<g1(512L * 2048), 256, 0, stream>>>(lw[li][14], bw_f2[li], 2048, 512, 512, 0);
    }
    for (int ci = 0; ci < 3; ++ci)
        cvt_convT_k<<<g1(512L * 512 * 3), 256, 0, stream>>>(conv_w[ci], bw_conv[ci], 512);
    cvtT_k<<<g1(512L * 512), 256, 0, stream>>>(mean_w1, bw_m1, 512, 512, 512, 0);
    cvtT_k<<<g1(128L * 512), 256, 0, stream>>>(mean_w2, bw_m2, 512, 32, 128, 0);
    cvtT_k<<<g1(512L * 512), 256, 0, stream>>>(alea_w1, bw_a1, 512, 512, 512, 0);
    cvtT_k<<<g1(128L * 512), 256, 0, stream>>>(alea_w2, bw_a2, 512, 32, 128, 0);

    pe_k<<<g1((long)T_ * D / 2), 256, 0, stream>>>(pe, T_, D);
    maskts_k<<<g1(R), 256, 0, stream>>>(M, mts, R, DI);
    buildz_k<<<g1((long)R * DI), 256, 0, stream>>>(X, M, z, R, DI);

    // ---- embedding: relu(z@W1+b1)@W2+b2, then missing-embed substitution ----
    launch_gemm(z, bw_e1, embed_b1, nullptr, t1, R, D, 2 * DI, GT_RELU, 0, 0, stream);
    launch_gemm(t1, bw_e2, embed_b2, nullptr, h, R, D, D, GT_NONE, 0, 0, stream);
    maskembed_k<<<g1(RD), 256, 0, stream>>>(h, mts, missing_embed, R);

    // ---- transformer layers ----
    for (int li = 0; li < L; ++li) {
        addpe_k<<<g1(RD), 256, 0, stream>>>(h, pe, xb, T_);
        launch_gemm(xb, bw_qkv[li], qb[li], nullptr, qkv, R, 1536, D, GT_NONE, 0, 0, stream);
        cvtqk_k<<<g1((long)R * 1024), 256, 0, stream>>>(qkv, qkbf, R);
        vtrans_k<<<g1((long)B_ * H * 64 * ATT_TPAD), 256, 0, stream>>>(qkv, vtbf, B_, T_, H);
        {
            dim3 ga(cdiv_i(T_, 16), B_ * H);
            attn_wmma<<<ga, 128, 0, stream>>>(qkbf, vtbf, lw[li][9], lw[li][8], mts,
                                              attn, B_, T_, H);
        }
        launch_gemm(attn, bw_o[li], lw[li][7], h, t1, R, D, D, GT_NONE, 0, 0, stream);
        layernorm_k<<<R, 256, 0, stream>>>(t1, lw[li][10], lw[li][11], h);
        launch_gemm(h, bw_f1[li], lw[li][13], nullptr, ff1, R, 2048, D, GT_GELU, 0, 0, stream);
        launch_gemm(ff1, bw_f2[li], lw[li][15], h, t1, R, D, 2048, GT_NONE, 0, 0, stream);
        layernorm_k<<<R, 256, 0, stream>>>(t1, lw[li][16], lw[li][17], h);
    }

    // ---- dilated conv stack (each conv = 3 shifted GEMMs, residual fused) ----
    copyf_k<<<g1(RD), 256, 0, stream>>>(h, hl, RD);
    float* cin = hl; float* cout = hl2;
    for (int ci = 0; ci < 3; ++ci) {
        int dil = 1 << ci;
        const bf16_t* wt = bw_conv[ci];
        launch_gemm(cin, wt + 0L * 512 * 512, conv_b[ci], cin,  cout, R, D, D, GT_NONE, -dil, T_, stream);
        launch_gemm(cin, wt + 1L * 512 * 512, nullptr,    cout, cout, R, D, D, GT_NONE,    0, T_, stream);
        launch_gemm(cin, wt + 2L * 512 * 512, nullptr,    cout, cout, R, D, D, GT_NONE, +dil, T_, stream);
        float* tmp = cin; cin = cout; cout = tmp;
    }
    add_k<<<g1(RD), 256, 0, stream>>>(h, cin, xb, RD);   // xb = final h

    // ---- output heads ----
    launch_gemm(xb, bw_m1, mean_b1, nullptr, t1, R, D, D, GT_RELU, 0, 0, stream);
    launch_gemm(t1, bw_m2, mean_b2, nullptr, meanb, R, DI, D, GT_NONE, 0, 0, stream);
    launch_gemm(xb, bw_a1, alea_b1, nullptr, t1, R, D, D, GT_RELU, 0, 0, stream);
    launch_gemm(t1, bw_a2, alea_b2, nullptr, aleab, R, DI, D, GT_SOFTPLUS, 0, 0, stream);

    // ---- sigma statistics + final assembly ----
    const int npix = R * DI;
    red1_k<<<g1(npix), 256, 0, stream>>>(meanb, X, M, sc, npix);
    red2_k<<<g1(npix), 256, 0, stream>>>(meanb, X, M, sc, npix);
    final_k<<<g1(npix), 256, 0, stream>>>(meanb, aleab, X, M, sc, out, npix);
}
